// MultiHeadAttention_60567628808536
// MI455X (gfx1250) — matrix operand order repaired and checked by reading, not compiled, not run on hardware
//
#include <hip/hip_runtime.h>
#include <hip/hip_bf16.h>

typedef _Float16 v8h  __attribute__((ext_vector_type(8)));
typedef _Float16 v16h __attribute__((ext_vector_type(16)));
typedef float    v8f  __attribute__((ext_vector_type(8)));
typedef unsigned int u32x4 __attribute__((ext_vector_type(4)));
typedef int          i32x4 __attribute__((ext_vector_type(4)));
typedef int          i32x8 __attribute__((ext_vector_type(8)));

#define BB 4
#define LL 4096
#define EE 512
#define HH 8
#define DD 64
#define MROWS (BB*LL)   // 16384

#if defined(__HIP_DEVICE_COMPILE__) && __has_builtin(__builtin_amdgcn_tensor_load_to_lds)
#define HAVE_TDM 1
#else
#define HAVE_TDM 0
#endif

// ---------------------------------------------------------------------------
// WMMA fragment helpers (wave32, v_wmma_f32_16x16x32_f16)
// A (16x32): lane m=lane&15, half=lane>>4; VGPRs0-3 = K[half*16..+7],
//            VGPRs4-7 = K[half*16+8..+7]  -> two contiguous 16B loads
// B (32x16): lane holds col n=lane&15; K contiguous per lane when the B
//            operand is stored transposed row-major ([n][k])
// C (16x16): lane=col n, VGPR r -> row m = r + 8*half
// ---------------------------------------------------------------------------
__device__ __forceinline__ v16h frag_cat(v8h lo, v8h hi) {
  v16h r;
#pragma unroll
  for (int i = 0; i < 8; ++i) { r[i] = lo[i]; r[i + 8] = hi[i]; }
  return r;
}

__device__ __forceinline__ v16h load_a(const _Float16* __restrict__ base, int ld,
                                       int k, int lane) {
  int m = lane & 15, hf = lane >> 4;
  const _Float16* p = base + (size_t)m * ld + k + hf * 16;
  return frag_cat(*(const v8h*)p, *(const v8h*)(p + 8));
}

__device__ __forceinline__ v16h load_bt(const _Float16* __restrict__ bt, int ld,
                                        int k, int lane) {
  int n = lane & 15, hf = lane >> 4;
  const _Float16* p = bt + (size_t)n * ld + k + hf * 16;
  return frag_cat(*(const v8h*)p, *(const v8h*)(p + 8));
}

// B fragment from an LDS panel laid out [128 rows][32 k] (row stride 64B)
__device__ __forceinline__ v16h load_b_lds(const _Float16* panel, int nrow,
                                           int lane) {
  int n = lane & 15, hf = lane >> 4;
  const _Float16* p = panel + (nrow + n) * 32 + hf * 16;
  return frag_cat(*(const v8h*)p, *(const v8h*)(p + 8));
}

__device__ __forceinline__ v8f wmma16(v16h a, v16h b, v8f c) {
  return __builtin_amdgcn_wmma_f32_16x16x32_f16(false, a, false, b,
                                                (short)0, c, false, false);
}

__device__ __forceinline__ float feature_map(float x) {
  return x > 0.0f ? x + 1.0f : __expf(x);   // elu(x) + 1
}

#if HAVE_TDM
// ---------------------------------------------------------------------------
// Tensor Data Mover: 2D f16 tile (tile_w x tile_h elements) from a row-major
// tensor (stride_elems between rows) into LDS at lds_byte_off.
// D# per CDNA5 ISA 8.3/8.4: group0 = {count/type/addr}, group1 = dims/strides.
// ---------------------------------------------------------------------------
__device__ __forceinline__ void tdm_load_tile_f16(
    const void* gaddr, unsigned lds_byte_off, unsigned tensor_w,
    unsigned tensor_h, unsigned stride_elems, unsigned tile_w,
    unsigned tile_h) {
  unsigned long long ga = (unsigned long long)gaddr;
  u32x4 g0;
  g0[0] = 1u;                                               // count=1 (valid)
  g0[1] = lds_byte_off;                                     // lds_addr
  g0[2] = (unsigned)ga;                                     // global_addr lo
  g0[3] = (unsigned)((ga >> 32) & 0x1FFFFFFu) | (2u << 30); // addr hi | type=2
  i32x8 g1;
  g1[0] = (int)(1u << 16);                                  // data_size = 2B
  g1[1] = (int)((tensor_w & 0xFFFFu) << 16);                // tensor_dim0 lo
  g1[2] = (int)((tensor_w >> 16) | ((tensor_h & 0xFFFFu) << 16));
  g1[3] = (int)((tensor_h >> 16) | (tile_w << 16));         // tile_dim0
  g1[4] = (int)(tile_h & 0xFFFFu);                          // tile_dim1
  g1[5] = (int)stride_elems;                                // dim0_stride lo
  g1[6] = 0;
  g1[7] = 0;
  i32x4 z4 = {0, 0, 0, 0};
#if __clang_major__ >= 23
  i32x8 z8 = {0, 0, 0, 0, 0, 0, 0, 0};
  __builtin_amdgcn_tensor_load_to_lds(g0, g1, z4, z4, z8, 0);
#else
  __builtin_amdgcn_tensor_load_to_lds(g0, g1, z4, z4, 0);
#endif
}

__device__ __forceinline__ unsigned lds_offset_of(const void* p) {
  // addrspace(3) generic pointers: low 32 bits are the LDS byte offset
  return (unsigned)(unsigned long long)p;
}
#endif  // HAVE_TDM

// ---------------------------------------------------------------------------
// fp32 -> fp16 conversion, 8 elements per thread
// ---------------------------------------------------------------------------
__global__ void cvt8_kernel(const float* __restrict__ src,
                            _Float16* __restrict__ dst, int n8) {
  int i = blockIdx.x * blockDim.x + threadIdx.x;
  if (i < n8) {
    const float4* s = (const float4*)src + (size_t)i * 2;
    float4 a = s[0], b = s[1];
    v8h o;
    o[0] = (_Float16)a.x; o[1] = (_Float16)a.y;
    o[2] = (_Float16)a.z; o[3] = (_Float16)a.w;
    o[4] = (_Float16)b.x; o[5] = (_Float16)b.y;
    o[6] = (_Float16)b.z; o[7] = (_Float16)b.w;
    *((v8h*)dst + i) = o;
  }
}

// ---------------------------------------------------------------------------
// Fused projection GEMM: P = X @ W.T + b, then
//   mode 0 (Q): elu+1, store row-major Qf[B*L][E]
//   mode 1 (K): elu+1, store transposed Kt[B][H][D][L]
//   mode 2 (V): /L,    store transposed Vt[B][H][D][L]
// block = 256 thr (8 waves), block tile 128x128, wave tile 32x64.
// B (weight) panel 128x32 staged through LDS by the TDM, double-buffered.
// ---------------------------------------------------------------------------
__global__ __launch_bounds__(256) void proj_kernel(
    const _Float16* __restrict__ qh, const _Float16* __restrict__ kh,
    const _Float16* __restrict__ vh, const _Float16* __restrict__ Wqh,
    const _Float16* __restrict__ Wkh, const _Float16* __restrict__ Wvh,
    const float* __restrict__ bq, const float* __restrict__ bk,
    const float* __restrict__ bv, _Float16* __restrict__ Qf,
    _Float16* __restrict__ Kt, _Float16* __restrict__ Vt) {
  const int mode = blockIdx.z;
  const _Float16* X = (mode == 0) ? qh : (mode == 1) ? kh : vh;
  const _Float16* W = (mode == 0) ? Wqh : (mode == 1) ? Wkh : Wvh;
  const float* bias = (mode == 0) ? bq : (mode == 1) ? bk : bv;

  const int lane = threadIdx.x & 31;
  const int w = threadIdx.x >> 5;
  const int Mbase = blockIdx.x * 128 + (w & 3) * 32;
  const int Nblock = blockIdx.y * 128;
  const int Nwave = (w >> 2) * 64;   // within block panel

  v8f acc[2][4] = {};

#if HAVE_TDM
  __shared__ _Float16 bpanel[2][128 * 32];   // 16 KB double buffer
  const _Float16* Wpanel = W + (size_t)Nblock * EE;
  if (w == 0)
    tdm_load_tile_f16(Wpanel, lds_offset_of(&bpanel[0][0]), EE, EE, EE, 32, 128);
  for (int k = 0; k < EE; k += 32) {
    const int cur = (k >> 5) & 1;
    __syncthreads();                     // readers of buf[cur^1] are done
    if (w == 0) {
      if (k + 32 < EE) {
        tdm_load_tile_f16(Wpanel + k + 32, lds_offset_of(&bpanel[cur ^ 1][0]),
                          EE, EE, EE, 32, 128);
        __builtin_amdgcn_s_wait_tensorcnt(1);   // current panel resident
      } else {
        __builtin_amdgcn_s_wait_tensorcnt(0);
      }
    }
    __syncthreads();                     // panel visible to all waves
    v16h a0 = load_a(X + (size_t)Mbase * EE, EE, k, lane);
    v16h a1 = load_a(X + (size_t)(Mbase + 16) * EE, EE, k, lane);
    v16h b[4];
#pragma unroll
    for (int j = 0; j < 4; ++j) b[j] = load_b_lds(bpanel[cur], Nwave + j * 16, lane);
#pragma unroll
    for (int j = 0; j < 4; ++j) {
      acc[0][j] = wmma16(a0, b[j], acc[0][j]);
      acc[1][j] = wmma16(a1, b[j], acc[1][j]);
    }
  }
#else
  for (int k = 0; k < EE; k += 32) {
    v16h a0 = load_a(X + (size_t)Mbase * EE, EE, k, lane);
    v16h a1 = load_a(X + (size_t)(Mbase + 16) * EE, EE, k, lane);
    v16h b[4];
#pragma unroll
    for (int j = 0; j < 4; ++j)
      b[j] = load_bt(W + (size_t)(Nblock + Nwave + j * 16) * EE, EE, k, lane);
#pragma unroll
    for (int j = 0; j < 4; ++j) {
      acc[0][j] = wmma16(a0, b[j], acc[0][j]);
      acc[1][j] = wmma16(a1, b[j], acc[1][j]);
    }
  }
#endif

  const int n = lane & 15, hf = lane >> 4;
#pragma unroll
  for (int i = 0; i < 2; ++i) {
#pragma unroll
    for (int j = 0; j < 4; ++j) {
      const int col = Nblock + Nwave + j * 16 + n;
      const float bb = bias[col];
      if (mode == 0) {
#pragma unroll
        for (int r = 0; r < 8; ++r) {
          int row = Mbase + i * 16 + 8 * hf + r;
          Qf[(size_t)row * EE + col] = (_Float16)feature_map(acc[i][j][r] + bb);
        }
      } else {
        _Float16* T = (mode == 1) ? Kt : Vt;
        const int h = col >> 6, d = col & 63;
        const int row0 = Mbase + i * 16 + 8 * hf;
        const int b_ = row0 >> 12;          // / LL
        const int l0 = row0 & (LL - 1);
        v8h pk;
#pragma unroll
        for (int r = 0; r < 8; ++r) {
          float val = acc[i][j][r] + bb;
          val = (mode == 1) ? feature_map(val) : val * (1.0f / LL);
          pk[r] = (_Float16)val;
        }
        *(v8h*)(T + (((size_t)b_ * HH + h) * DD + d) * LL + l0) = pk;
      }
    }
  }
}

// ---------------------------------------------------------------------------
// Ksum[b,h,d] = sum_s Kt[b,h,d,s]
// ---------------------------------------------------------------------------
__global__ void ksum_kernel(const _Float16* __restrict__ Kt,
                            float* __restrict__ Ksum) {
  int idx = blockIdx.x * blockDim.x + threadIdx.x;  // B*H*D = 2048
  const _Float16* p = Kt + (size_t)idx * LL;
  float s = 0.0f;
  for (int i = 0; i < LL; i += 8) {
    v8h v = *(const v8h*)(p + i);
#pragma unroll
    for (int j = 0; j < 8; ++j) s += (float)v[j];
  }
  Ksum[idx] = s;
}

// ---------------------------------------------------------------------------
// KV = Kt @ Vt.T per (b,h): [64 x 4096] x [4096 x 64]; stored transposed KVt.
// 8 waves: 4 d-slabs x 2 sequence-halves, LDS reduction across halves.
// ---------------------------------------------------------------------------
__global__ __launch_bounds__(256) void kv_kernel(const _Float16* __restrict__ Kt,
                                                 const _Float16* __restrict__ Vt,
                                                 _Float16* __restrict__ KVt) {
  __shared__ float red[4][4][8][32];
  const int lane = threadIdx.x & 31;
  const int w = threadIdx.x >> 5;
  const int bh = blockIdx.x;
  const int ds = (w & 3) * 16;
  const int sh = w >> 2;

  const _Float16* A = Kt + (size_t)bh * DD * LL + (size_t)ds * LL;
  const _Float16* Bt = Vt + (size_t)bh * DD * LL;

  v8f acc[4] = {};
  const int s0 = sh * (LL / 2);
  for (int s = s0; s < s0 + LL / 2; s += 32) {
    v16h a = load_a(A, LL, s, lane);
#pragma unroll
    for (int j = 0; j < 4; ++j) {
      v16h b = load_bt(Bt + (size_t)(j * 16) * LL, LL, s, lane);
      acc[j] = wmma16(a, b, acc[j]);
    }
  }

  if (sh == 1) {
#pragma unroll
    for (int j = 0; j < 4; ++j)
#pragma unroll
      for (int r = 0; r < 8; ++r) red[w - 4][j][r][lane] = acc[j][r];
  }
  __syncthreads();
  if (sh == 0) {
    const int n = lane & 15, hf = lane >> 4;
#pragma unroll
    for (int j = 0; j < 4; ++j) {
      v8h pk;
#pragma unroll
      for (int r = 0; r < 8; ++r)
        pk[r] = (_Float16)(acc[j][r] + red[w][j][r][lane]);
      *(v8h*)(KVt + (size_t)bh * DD * DD + (size_t)(j * 16 + n) * DD + ds +
              8 * hf) = pk;
    }
  }
}

// ---------------------------------------------------------------------------
// Z[b,l,h] = 1 / (Qf[b,l,h,:] . Ksum[b,h,:] + eps)
// ---------------------------------------------------------------------------
__global__ void z_kernel(const _Float16* __restrict__ Qf,
                         const float* __restrict__ Ksum,
                         float* __restrict__ Z) {
  int idx = blockIdx.x * blockDim.x + threadIdx.x;  // B*L*H
  int h = idx & (HH - 1);
  int bl = idx >> 3;
  const _Float16* q = Qf + (size_t)bl * EE + h * DD;
  int b_ = bl >> 12;
  const float* ks = Ksum + ((size_t)b_ * HH + h) * DD;
  float s = 0.0f;
#pragma unroll 8
  for (int d = 0; d < DD; ++d) s += (float)q[d] * ks[d];
  Z[idx] = 1.0f / (s + 1e-6f);
}

// ---------------------------------------------------------------------------
// O[b,l,h,:] = (Qf[b,l,h,:] @ KV[b,h]) * Z[b,l,h] * L
// ---------------------------------------------------------------------------
__global__ __launch_bounds__(256) void qkv_kernel(
    const _Float16* __restrict__ Qf, const _Float16* __restrict__ KVt,
    const float* __restrict__ Z, _Float16* __restrict__ O) {
  const int lane = threadIdx.x & 31;
  const int w = threadIdx.x >> 5;
  const int bh = blockIdx.y;
  const int b_ = bh >> 3, h = bh & 7;
  const int lbase = blockIdx.x * 256 + w * 32;

  const _Float16* A = Qf + ((size_t)b_ * LL + lbase) * EE + h * DD;
  const _Float16* Bt = KVt + (size_t)bh * DD * DD;

  v8f acc[2][4] = {};
#pragma unroll
  for (int k = 0; k < DD; k += 32) {
    v16h a0 = load_a(A, EE, k, lane);
    v16h a1 = load_a(A + (size_t)16 * EE, EE, k, lane);
#pragma unroll
    for (int j = 0; j < 4; ++j) {
      v16h b = load_bt(Bt + (size_t)(j * 16) * DD, DD, k, lane);
      acc[0][j] = wmma16(a0, b, acc[0][j]);
      acc[1][j] = wmma16(a1, b, acc[1][j]);
    }
  }

  const int n = lane & 15, hf = lane >> 4;
#pragma unroll
  for (int i = 0; i < 2; ++i) {
#pragma unroll
    for (int r = 0; r < 8; ++r) {
      const int l = lbase + i * 16 + 8 * hf + r;
      const float z = Z[((size_t)b_ * LL + l) * HH + h] * (float)LL;
#pragma unroll
      for (int j = 0; j < 4; ++j) {
        O[((size_t)b_ * LL + l) * EE + h * DD + j * 16 + n] =
            (_Float16)(acc[i][j][r] * z);
      }
    }
  }
}

// ---------------------------------------------------------------------------
// message = O @ Wm.T -> f32 output; TDM-staged weight panels as in proj.
// ---------------------------------------------------------------------------
__global__ __launch_bounds__(256) void merge_kernel(
    const _Float16* __restrict__ O, const _Float16* __restrict__ Wmh,
    float* __restrict__ out) {
  const int lane = threadIdx.x & 31;
  const int w = threadIdx.x >> 5;
  const int Mbase = blockIdx.x * 128 + (w & 3) * 32;
  const int Nblock = blockIdx.y * 128;
  const int Nwave = (w >> 2) * 64;

  v8f acc[2][4] = {};

#if HAVE_TDM
  __shared__ _Float16 bpanel[2][128 * 32];
  const _Float16* Wpanel = Wmh + (size_t)Nblock * EE;
  if (w == 0)
    tdm_load_tile_f16(Wpanel, lds_offset_of(&bpanel[0][0]), EE, EE, EE, 32, 128);
  for (int k = 0; k < EE; k += 32) {
    const int cur = (k >> 5) & 1;
    __syncthreads();
    if (w == 0) {
      if (k + 32 < EE) {
        tdm_load_tile_f16(Wpanel + k + 32, lds_offset_of(&bpanel[cur ^ 1][0]),
                          EE, EE, EE, 32, 128);
        __builtin_amdgcn_s_wait_tensorcnt(1);
      } else {
        __builtin_amdgcn_s_wait_tensorcnt(0);
      }
    }
    __syncthreads();
    v16h a0 = load_a(O + (size_t)Mbase * EE, EE, k, lane);
    v16h a1 = load_a(O + (size_t)(Mbase + 16) * EE, EE, k, lane);
    v16h b[4];
#pragma unroll
    for (int j = 0; j < 4; ++j) b[j] = load_b_lds(bpanel[cur], Nwave + j * 16, lane);
#pragma unroll
    for (int j = 0; j < 4; ++j) {
      acc[0][j] = wmma16(a0, b[j], acc[0][j]);
      acc[1][j] = wmma16(a1, b[j], acc[1][j]);
    }
  }
#else
  for (int k = 0; k < EE; k += 32) {
    v16h a0 = load_a(O + (size_t)Mbase * EE, EE, k, lane);
    v16h a1 = load_a(O + (size_t)(Mbase + 16) * EE, EE, k, lane);
    v16h b[4];
#pragma unroll
    for (int j = 0; j < 4; ++j)
      b[j] = load_bt(Wmh + (size_t)(Nblock + Nwave + j * 16) * EE, EE, k, lane);
#pragma unroll
    for (int j = 0; j < 4; ++j) {
      acc[0][j] = wmma16(a0, b[j], acc[0][j]);
      acc[1][j] = wmma16(a1, b[j], acc[1][j]);
    }
  }
#endif

  const int n = lane & 15, hf = lane >> 4;
#pragma unroll
  for (int i = 0; i < 2; ++i)
#pragma unroll
    for (int j = 0; j < 4; ++j)
#pragma unroll
      for (int r = 0; r < 8; ++r) {
        int row = Mbase + i * 16 + 8 * hf + r;
        out[(size_t)row * EE + Nblock + Nwave + j * 16 + n] = acc[i][j][r];
      }
}

// ---------------------------------------------------------------------------
extern "C" void kernel_launch(void* const* d_in, const int* in_sizes, int n_in,
                              void* d_out, int out_size, void* d_ws,
                              size_t ws_size, hipStream_t stream) {
  const float* q = (const float*)d_in[0];
  const float* k = (const float*)d_in[1];
  const float* v = (const float*)d_in[2];
  const float* Wq = (const float*)d_in[3];
  const float* bq = (const float*)d_in[4];
  const float* Wk = (const float*)d_in[5];
  const float* bk = (const float*)d_in[6];
  const float* Wv = (const float*)d_in[7];
  const float* bv = (const float*)d_in[8];
  const float* Wm = (const float*)d_in[9];
  float* out = (float*)d_out;

  const size_t NX = (size_t)MROWS * EE;   // 8,388,608 elements
  const size_t NW = (size_t)EE * EE;      // 262,144 elements

  char* ws = (char*)d_ws;
  size_t off = 0;
  auto take = [&](size_t bytes) -> char* {
    char* p = ws + off;
    off += (bytes + 255) & ~(size_t)255;
    return p;
  };
  _Float16* qh  = (_Float16*)take(NX * 2);
  _Float16* kh  = (_Float16*)take(NX * 2);
  _Float16* vh  = (_Float16*)take(NX * 2);
  _Float16* Wqh = (_Float16*)take(NW * 2);
  _Float16* Wkh = (_Float16*)take(NW * 2);
  _Float16* Wvh = (_Float16*)take(NW * 2);
  _Float16* Wmh = (_Float16*)take(NW * 2);
  _Float16* Qf  = (_Float16*)take(NX * 2);
  _Float16* Kt  = (_Float16*)take(NX * 2);
  _Float16* Vt  = (_Float16*)take(NX * 2);
  _Float16* Ob  = (_Float16*)take(NX * 2);
  _Float16* KVt = (_Float16*)take((size_t)BB * HH * DD * DD * 2);
  float* Ksum   = (float*)take((size_t)BB * HH * DD * 4);
  float* Zb     = (float*)take((size_t)BB * LL * HH * 4);

  // fp32 -> fp16 conversions (8 elems / thread)
  cvt8_kernel<<<(int)(NX / 8 / 256), 256, 0, stream>>>(q, qh, (int)(NX / 8));
  cvt8_kernel<<<(int)(NX / 8 / 256), 256, 0, stream>>>(k, kh, (int)(NX / 8));
  cvt8_kernel<<<(int)(NX / 8 / 256), 256, 0, stream>>>(v, vh, (int)(NX / 8));
  cvt8_kernel<<<(int)(NW / 8 / 256), 256, 0, stream>>>(Wq, Wqh, (int)(NW / 8));
  cvt8_kernel<<<(int)(NW / 8 / 256), 256, 0, stream>>>(Wk, Wkh, (int)(NW / 8));
  cvt8_kernel<<<(int)(NW / 8 / 256), 256, 0, stream>>>(Wv, Wvh, (int)(NW / 8));
  cvt8_kernel<<<(int)(NW / 8 / 256), 256, 0, stream>>>(Wm, Wmh, (int)(NW / 8));

  // fused projections + feature map (+ transposed K/V stores)
  proj_kernel<<<dim3(MROWS / 128, EE / 128, 3), 256, 0, stream>>>(
      qh, kh, vh, Wqh, Wkh, Wvh, bq, bk, bv, Qf, Kt, Vt);

  // K row sums
  ksum_kernel<<<(BB * HH * DD) / 256, 256, 0, stream>>>(Kt, Ksum);

  // KV = K^T V (per batch-head), transposed output
  kv_kernel<<<BB * HH, 256, 0, stream>>>(Kt, Vt, KVt);

  // normalizers
  z_kernel<<<(BB * LL * HH) / 256, 256, 0, stream>>>(Qf, Ksum, Zb);

  // O = (Q @ KV) * Z * L
  qkv_kernel<<<dim3(LL / 256, BB * HH), 256, 0, stream>>>(Qf, KVt, Zb, Ob);

  // message = O @ Wm.T
  merge_kernel<<<dim3(MROWS / 128, EE / 128), 256, 0, stream>>>(Ob, Wmh, out);

  (void)in_sizes; (void)n_in; (void)out_size; (void)ws_size;
}
